// GRUFn_5901285065263
// MI455X (gfx1250) — matrix operand order repaired and checked by reading, not compiled, not run on hardware
//
#include <hip/hip_runtime.h>
#include <hip/hip_bf16.h>

// ---------------------------------------------------------------------------
// GRU, T=32768, I=512, H=512.
// Phase 1: igates = xs @ W_ih^T + bias  (bf16 WMMA, fp32 accumulate) -> ws
// Phase 2: sequential scan, W_hh register-resident across 16 persistent WGs,
//          h exchanged through L2 with a device-scope atomic grid barrier.
// ---------------------------------------------------------------------------

#define T_STEPS 32768
#define I_DIM   512
#define H_DIM   512
#define G3H     1536          // 3*H
#define GBLK    16            // persistent blocks in the scan

typedef unsigned short u16;
typedef __attribute__((ext_vector_type(16))) __bf16 v16bf;
typedef __attribute__((ext_vector_type(8)))  float  v8f;

union Frag { v16bf v; uint4 q[2]; };

// ---- fp32 -> bf16 (round to nearest even) ---------------------------------
__device__ inline u16 f2bf(float f) {
  unsigned u = __float_as_uint(f);
  u += 0x7FFFu + ((u >> 16) & 1u);
  return (u16)(u >> 16);
}

__global__ void k_f2bf4(const float4* __restrict__ in, ushort4* __restrict__ out, int n4) {
  int i = blockIdx.x * blockDim.x + threadIdx.x;
  if (i < n4) {
    float4 f = in[i];
    ushort4 o;
    o.x = f2bf(f.x); o.y = f2bf(f.y); o.z = f2bf(f.z); o.w = f2bf(f.w);
    out[i] = o;
  }
}

// ---- init: zero grid-barrier counter and h double buffer ------------------
__global__ void k_init(float* hbuf, unsigned* cnt) {
  if (threadIdx.x == 0) *cnt = 0u;
  if (threadIdx.x < 2 * H_DIM) hbuf[threadIdx.x] = 0.f;   // h0 = 0
}

// ---------------------------------------------------------------------------
// Phase 1: WMMA GEMM. C(T,1536) = A(T,512) * B(512,1536) + bias, bf16 inputs.
// One wave computes a 16x64 strip (4 column tiles), k in steps of 32.
// Fragments are built straight from global memory using the same contiguous-k
// per-lane order for A and B (any shared k order is valid):
//   A 16x32 bf16: lane L holds row M=L%16; halves 0..7 = K [k0+16g, +8),
//                 halves 8..15 = K [k0+16g+8, +8), g = L/16.
//   B 32x16 bf16: lane L holds col N=L%16; 16 contiguous K at k0+16g.
//                 B[k,n] = W_ih[n,k] so this is a contiguous row of W_ih.
//   C/D v8f:      VGPR r, lane L -> M = r + 8g, N = L%16.
// ---------------------------------------------------------------------------
__global__ __launch_bounds__(256) void k_igemm(const u16* __restrict__ xbf,
                                               const u16* __restrict__ wbf,
                                               const float* __restrict__ bias,
                                               float* __restrict__ ig) {
  const int lane   = threadIdx.x & 31;
  const int lane16 = lane & 15;
  const int grp    = lane >> 4;
  const int wave   = blockIdx.x * (blockDim.x >> 5) + (threadIdx.x >> 5);
  const int mt     = wave / 24;          // 0..2047  (row tile of 16)
  const int ns     = wave - mt * 24;     // 0..23    (column strip of 64)
  const int m      = mt * 16 + lane16;   // global A row for this lane
  const int nbase  = ns * 64;

  v8f acc[4];
#pragma unroll
  for (int i = 0; i < 4; ++i)
#pragma unroll
    for (int r = 0; r < 8; ++r) acc[i][r] = 0.f;

  const u16* arow = xbf + (size_t)m * I_DIM;

  for (int k0 = 0; k0 < I_DIM; k0 += 32) {
    Frag a;
    a.q[0] = *(const uint4*)(arow + k0 + 16 * grp);       // K [k0+16g, +8) (8 bf16)
    a.q[1] = *(const uint4*)(arow + k0 + 16 * grp + 8);   // K [k0+16g+8, +8) (8 bf16)
#pragma unroll
    for (int nt = 0; nt < 4; ++nt) {
      const u16* brow = wbf + (size_t)(nbase + nt * 16 + lane16) * I_DIM + k0 + 16 * grp;
      Frag b;
      b.q[0] = *(const uint4*)(brow);
      b.q[1] = *(const uint4*)(brow + 8);
      acc[nt] = __builtin_amdgcn_wmma_f32_16x16x32_bf16(
          /*neg_a=*/false, a.v, /*neg_b=*/false, b.v,
          /*c_mod=*/(short)0, acc[nt], /*reuse_a=*/false, /*reuse_b=*/false);
    }
  }

#pragma unroll
  for (int nt = 0; nt < 4; ++nt) {
    const int n  = nbase + nt * 16 + lane16;
    const float bv = bias[n];
#pragma unroll
    for (int r = 0; r < 8; ++r) {
      const int M = mt * 16 + r + 8 * grp;
      ig[(size_t)M * G3H + n] = acc[nt][r] + bv;
    }
  }
}

// ---- fast elementwise ------------------------------------------------------
__device__ inline float sigmoid_(float x) { return 1.f / (1.f + __expf(-x)); }
__device__ inline float tanh_(float x)    { return 1.f - 2.f / (__expf(2.f * x) + 1.f); }

// ---------------------------------------------------------------------------
// Phase 2: sequential scan. 16 blocks x 1024 threads (32 waves). Wave w of
// block g owns hidden unit j = g*32+w and rows {j, 512+j, 1024+j} of W_hh,
// kept in 48 VGPRs/lane:  wX[i].{x..w} = W[row, 128*i + 4*lane + 0..3].
// Matching h layout: float4 loads of the LDS h copy at index lane + 32*i.
// ---------------------------------------------------------------------------
__global__ __launch_bounds__(1024) void k_gru_scan(const float* __restrict__ whh,
                                                   const float* __restrict__ ig,
                                                   const float* __restrict__ bias_n,
                                                   float* __restrict__ out,
                                                   float* hbuf, unsigned* cnt) {
  __shared__ float s_h[H_DIM];
  const int tid  = threadIdx.x;
  const int lane = tid & 31;
  const int w    = tid >> 5;
  const int j    = blockIdx.x * 32 + w;      // hidden unit 0..511

  // Load this wave's 3 weight rows into registers (done once, stays resident).
  float4 wr[4], wz[4], wn[4];
  {
    const float4* R = (const float4*)(whh + (size_t)(j)            * H_DIM);
    const float4* Z = (const float4*)(whh + (size_t)(H_DIM + j)    * H_DIM);
    const float4* N = (const float4*)(whh + (size_t)(2 * H_DIM + j) * H_DIM);
#pragma unroll
    for (int i = 0; i < 4; ++i) {
      wr[i] = R[lane + 32 * i];
      wz[i] = Z[lane + 32 * i];
      wn[i] = N[lane + 32 * i];
    }
  }
  const float bn = bias_n[j];

  for (int t = 0; t < T_STEPS; ++t) {
    float* hsrc = hbuf + (t & 1) * H_DIM;
    float* hdst = hbuf + ((t + 1) & 1) * H_DIM;

    // Stage current h into LDS (device-scope loads -> fresh data from L2).
    if (tid < H_DIM)
      s_h[tid] = __hip_atomic_load(hsrc + tid, __ATOMIC_RELAXED, __HIP_MEMORY_SCOPE_AGENT);
    __syncthreads();

    // Three 512-length dot products per wave, weights in registers.
    const float4* h4 = (const float4*)s_h;
    float ar = 0.f, az = 0.f, an = 0.f;
#pragma unroll
    for (int i = 0; i < 4; ++i) {
      float4 h = h4[lane + 32 * i];
      ar = fmaf(wr[i].x, h.x, fmaf(wr[i].y, h.y, fmaf(wr[i].z, h.z, fmaf(wr[i].w, h.w, ar))));
      az = fmaf(wz[i].x, h.x, fmaf(wz[i].y, h.y, fmaf(wz[i].z, h.z, fmaf(wz[i].w, h.w, az))));
      an = fmaf(wn[i].x, h.x, fmaf(wn[i].y, h.y, fmaf(wn[i].z, h.z, fmaf(wn[i].w, h.w, an))));
    }
#pragma unroll
    for (int off = 16; off > 0; off >>= 1) {
      ar += __shfl_xor(ar, off, 32);
      az += __shfl_xor(az, off, 32);
      an += __shfl_xor(an, off, 32);
    }

    if (lane == 0) {
      const float* igt = ig + (size_t)t * G3H + j;
      float r = sigmoid_(igt[0]        + ar);
      float z = sigmoid_(igt[H_DIM]    + az);
      float n = tanh_(   igt[2 * H_DIM] + r * (an + bn));
      float hj   = s_h[j];
      float hnew = n + z * (hj - n);
      out[(size_t)t * H_DIM + j] = hnew;
      __hip_atomic_store(hdst + j, hnew, __ATOMIC_RELAXED, __HIP_MEMORY_SCOPE_AGENT);
      __threadfence();   // make h visible device-wide before signaling
      if (t + 1 < T_STEPS)
        __builtin_prefetch(ig + (size_t)(t + 1) * G3H + j, 0, 1);  // global_prefetch_b8
    }
    __syncthreads();

    // Grid barrier: monotonic counter, one signaler/spinner per block.
    if (tid == 0) {
      __hip_atomic_fetch_add(cnt, 1u, __ATOMIC_RELEASE, __HIP_MEMORY_SCOPE_AGENT);
      const unsigned target = (unsigned)(t + 1) * GBLK;
      while (__hip_atomic_load(cnt, __ATOMIC_ACQUIRE, __HIP_MEMORY_SCOPE_AGENT) < target)
        __builtin_amdgcn_s_sleep(1);
    }
    __syncthreads();
  }
}

// ---------------------------------------------------------------------------
extern "C" void kernel_launch(void* const* d_in, const int* in_sizes, int n_in,
                              void* d_out, int out_size, void* d_ws, size_t ws_size,
                              hipStream_t stream) {
  (void)in_sizes; (void)n_in; (void)out_size; (void)ws_size;

  const float* xs   = (const float*)d_in[0];   // (T, I)
  const float* wih  = (const float*)d_in[1];   // (3H, I)
  const float* whh  = (const float*)d_in[2];   // (3H, H)
  const float* bias = (const float*)d_in[3];   // (3H,)
  const float* bn   = (const float*)d_in[4];   // (H,)
  float* out = (float*)d_out;                  // (T, H)

  // Workspace layout (bytes):
  //   [0)        grid-barrier counter
  //   [1024)     h double buffer: 2*512 floats
  //   [8192)     xs in bf16:   T*I u16      = 33,554,432 B
  //   [+33.5MB)  W_ih in bf16: 3H*I u16    =  1,572,864 B
  //   [+35.1MB)  igates fp32:  T*3H floats = 201,326,592 B
  char* ws = (char*)d_ws;
  unsigned* cnt = (unsigned*)ws;
  float*    hbuf = (float*)(ws + 1024);
  u16*      xbf  = (u16*)(ws + 8192);
  u16*      wbf  = (u16*)(ws + 8192 + (size_t)T_STEPS * I_DIM * 2);
  float*    ig   = (float*)(ws + 8192 + (size_t)T_STEPS * I_DIM * 2
                                      + (size_t)G3H * I_DIM * 2);

  k_init<<<1, 1024, 0, stream>>>(hbuf, cnt);

  const int nx4 = T_STEPS * I_DIM / 4;
  const int nw4 = G3H * I_DIM / 4;
  k_f2bf4<<<(nx4 + 255) / 256, 256, 0, stream>>>((const float4*)xs, (ushort4*)xbf, nx4);
  k_f2bf4<<<(nw4 + 255) / 256, 256, 0, stream>>>((const float4*)wih, (ushort4*)wbf, nw4);

  // jobs = (T/16 row tiles) * (1536/64 col strips) = 2048*24 = 49152 waves,
  // 8 waves/block -> 6144 blocks.
  k_igemm<<<6144, 256, 0, stream>>>(xbf, wbf, bias, ig);

  k_gru_scan<<<GBLK, 1024, 0, stream>>>(whh, ig, bn, out, hbuf, cnt);
}
